// Node_readout_86122684220175
// MI455X (gfx1250) — matrix operand order repaired and checked by reading, not compiled, not run on hardware
//
#include <hip/hip_runtime.h>

typedef _Float16 v16h __attribute__((ext_vector_type(16)));
typedef _Float16 v8h  __attribute__((ext_vector_type(8)));
typedef float    v8f  __attribute__((ext_vector_type(8)));

#define NB    512   // batch (graphs)
#define MAXA  256
#define MAXP  64
#define DIM   256
#define NH    8
#define DKH   32

// ---------------------------------------------------------------------------
// WMMA helpers (CDNA5 v_wmma_f32_16x16x32_f16)
// ---------------------------------------------------------------------------
__device__ __forceinline__ v8f wmma16x16x32(v16h a, v16h b, v8f c) {
  return __builtin_amdgcn_wmma_f32_16x16x32_f16(false, a, false, b, (short)0, c,
                                                false, false);
}

// A-matrix fragment (16 M x 32 K, f16), source row-major with leading dim ld.
// ISA layout: lanes 0-15: M=lane, K={0..7, 16..23}; lanes 16-31: M=lane-16,
// K={8..15, 24..31}. Two 16-byte loads per lane.
__device__ __forceinline__ v16h load_fragA(const _Float16* __restrict__ base, int ld) {
  const int lane = threadIdx.x & 31;
  const int row  = lane & 15;
  const int ko   = (lane >> 4) << 3;          // 0 or 8
  const _Float16* p = base + (size_t)row * ld + ko;
  union { v16h f; v8h h[2]; } u;
  u.h[0] = *reinterpret_cast<const v8h*>(p);        // K = ko .. ko+7
  u.h[1] = *reinterpret_cast<const v8h*>(p + 16);   // K = 16+ko .. 16+ko+7
  return u.f;
}

// B-matrix fragment (32 K x 16 N, f16), source stored TRANSPOSED: Bt[N][K]
// row-major with leading dim ld. ISA layout: lanes 0-15: N=lane, K={0..7, 16..23};
// lanes 16-31: N=lane-16, K={8..15, 24..31}.
__device__ __forceinline__ v16h load_fragB(const _Float16* __restrict__ base, int ld) {
  const int lane = threadIdx.x & 31;
  const int n  = lane & 15;
  const int ko = (lane >> 4) << 3;            // 0 or 8
  const _Float16* p = base + (size_t)n * ld + ko;
  union { v16h f; v8h h[2]; } u;
  u.h[0] = *reinterpret_cast<const v8h*>(p);
  u.h[1] = *reinterpret_cast<const v8h*>(p + 16);
  return u.f;
}

// ---------------------------------------------------------------------------
// K0: prefix sums of jagged sizes
// ---------------------------------------------------------------------------
__global__ void k_prefix(const int* __restrict__ a_sizes, const int* __restrict__ p_sizes,
                         int* __restrict__ starts_a, int* __restrict__ starts_p) {
  if (threadIdx.x == 0 && blockIdx.x == 0) {
    int sa = 0, sp = 0;
    for (int i = 0; i < NB; ++i) {
      starts_a[i] = sa; sa += a_sizes[i];
      starts_p[i] = sp; sp += p_sizes[i];
    }
  }
}

// K1: weight transpose + f32 -> f16 (Wt[n][k] = W[k][n])
__global__ void k_wt(const float* __restrict__ W, _Float16* __restrict__ Wt) {
  int idx = blockIdx.x * blockDim.x + threadIdx.x;   // 256*256
  int n = idx >> 8, k = idx & 255;
  Wt[(size_t)n * DIM + k] = (_Float16)W[(size_t)k * DIM + n];
}

// K2: jagged gather -> zero-padded f16 [NB*rows, DIM]
__global__ void k_gather(const float* __restrict__ flat, const int* __restrict__ sizes,
                         const int* __restrict__ starts, _Float16* __restrict__ out,
                         int rows_per_graph) {
  int row = blockIdx.x;
  int b = row / rows_per_graph;
  int r = row - b * rows_per_graph;
  bool valid = r < sizes[b];
  const float* sp = flat + (size_t)(starts[b] + r) * DIM;
  _Float16* op = out + (size_t)row * DIM;
  for (int c = threadIdx.x; c < DIM; c += blockDim.x) {
    float v = valid ? sp[c] : 0.0f;
    op[c] = (_Float16)v;
  }
}

// K3: C[M,256] (f16) = A[M,256] (f16) x Bt[256][256] (f16, N-major) + bias
// Block tile 128M x 64N; 8 waves, each wave 16M x 64N (4 accumulators).
__global__ __launch_bounds__(256) void k_gemm_f16out(
    const _Float16* __restrict__ A, const _Float16* __restrict__ Bt,
    const float* __restrict__ bias, _Float16* __restrict__ C, int M) {
  const int wave = threadIdx.x >> 5;
  const int lane = threadIdx.x & 31;
  const int m0 = blockIdx.x * 128 + wave * 16;
  const int n0 = blockIdx.y * 64;
  if (m0 >= M) return;                                  // wave-uniform
  v8f acc[4];
#pragma unroll
  for (int t = 0; t < 4; ++t) {
    float bv = bias[n0 + t * 16 + (lane & 15)];
#pragma unroll
    for (int r = 0; r < 8; ++r) acc[t][r] = bv;
  }
  for (int k0 = 0; k0 < DIM; k0 += 32) {
    v16h af = load_fragA(A + (size_t)m0 * DIM + k0, DIM);
#pragma unroll
    for (int t = 0; t < 4; ++t) {
      v16h bf = load_fragB(Bt + (size_t)(n0 + t * 16) * DIM + k0, DIM);
      acc[t] = wmma16x16x32(af, bf, acc[t]);
    }
  }
  const int colBase = lane & 15;
  const int rowAdd  = (lane >> 4) << 3;                 // C layout: M = r (+8)
#pragma unroll
  for (int t = 0; t < 4; ++t)
#pragma unroll
    for (int r = 0; r < 8; ++r)
      C[(size_t)(m0 + rowAdd + r) * DIM + (n0 + t * 16 + colBase)] = (_Float16)acc[t][r];
}

// K4: Vt[b][h][dk][key] = V[b*64+key][h*32+dk]  (B-operand layout for attn@V)
__global__ void k_vt(const _Float16* __restrict__ V, _Float16* __restrict__ Vt) {
  int idx = blockIdx.x * blockDim.x + threadIdx.x;      // NB*64*256
  int d   = idx & 255;
  int key = (idx >> 8) & 63;
  int b   = idx >> 14;
  int h = d >> 5, dk = d & 31;
  Vt[(((size_t)b * NH + h) * DKH + dk) * MAXP + key] = V[idx];
}

// K5: fused attention per (graph, head): scores -> mask/softmax -> attn@V
__global__ __launch_bounds__(256) void k_attn(
    const _Float16* __restrict__ Q,     // [NB*256, 256]
    const _Float16* __restrict__ Km,    // [NB*64, 256]
    const _Float16* __restrict__ Vt,    // [NB][H][32][64]
    const int* __restrict__ a_sizes, const int* __restrict__ p_sizes,
    _Float16* __restrict__ X) {         // [NB*256, 256]
  __shared__ __align__(16) _Float16 probs[8][16][MAXP]; // per-wave 16x64 tile
  const int b = blockIdx.x >> 3;
  const int h = blockIdx.x & 7;
  const int wave = threadIdx.x >> 5;
  const int lane = threadIdx.x & 31;
  const int na = a_sizes[b];
  const int np = p_sizes[b];
  const float scale = 0.1767766952966369f;              // 1/sqrt(32)
  const _Float16* Qb = Q  + (size_t)b * MAXA * DIM + h * DKH;
  const _Float16* Kb = Km + (size_t)b * MAXP * DIM + h * DKH;
  const _Float16* Vb = Vt + ((size_t)b * NH + h) * DKH * MAXP;
  const int colBase = lane & 15;
  const int rowAdd  = (lane >> 4) << 3;

  for (int mt = 0; mt < 2; ++mt) {
    const int q0 = wave * 32 + mt * 16;                 // 16-query tile
    // ---- scores = Qh (16x32) @ Kh^T (32x64), one K-step ----
    v8f s[4];
#pragma unroll
    for (int t = 0; t < 4; ++t)
#pragma unroll
      for (int r = 0; r < 8; ++r) s[t][r] = 0.0f;
    v16h qf = load_fragA(Qb + (size_t)q0 * DIM, DIM);
#pragma unroll
    for (int t = 0; t < 4; ++t) {
      v16h kf = load_fragB(Kb + (size_t)(t * 16) * DIM, DIM);
      s[t] = wmma16x16x32(qf, kf, s[t]);
    }
    // ---- mask + row softmax (row lives in 16 lanes of a half-wave) ----
#pragma unroll
    for (int r = 0; r < 8; ++r) {
      const int q = q0 + rowAdd + r;
      const bool qv = q < na;
      float v[4], mx = -3.0e38f;
#pragma unroll
      for (int t = 0; t < 4; ++t) {
        int key = t * 16 + colBase;
        bool kv = qv && (key < np);
        v[t] = kv ? s[t][r] * scale : -1.0e9f;
        mx = fmaxf(mx, v[t]);
      }
#pragma unroll
      for (int m = 1; m < 16; m <<= 1) mx = fmaxf(mx, __shfl_xor(mx, m, 32));
      float sum = 0.0f;
#pragma unroll
      for (int t = 0; t < 4; ++t) { v[t] = __expf(v[t] - mx); sum += v[t]; }
#pragma unroll
      for (int m = 1; m < 16; m <<= 1) sum += __shfl_xor(sum, m, 32);
      float inv = 1.0f / sum;
#pragma unroll
      for (int t = 0; t < 4; ++t)
        probs[wave][rowAdd + r][t * 16 + colBase] = (_Float16)(v[t] * inv);
    }
    // ---- X tile = probs (16x64) @ Vh (64x32); Vt supplies B-operand ----
    v8f x[2];
#pragma unroll
    for (int t = 0; t < 2; ++t)
#pragma unroll
      for (int r = 0; r < 8; ++r) x[t][r] = 0.0f;
    for (int k0 = 0; k0 < MAXP; k0 += 32) {             // wave-local LDS, in-order DS
      v16h pf = load_fragA(&probs[wave][0][0] + k0, MAXP);
#pragma unroll
      for (int t = 0; t < 2; ++t) {
        v16h vf = load_fragB(Vb + (size_t)(t * 16) * MAXP + k0, MAXP);
        x[t] = wmma16x16x32(pf, vf, x[t]);
      }
    }
#pragma unroll
    for (int t = 0; t < 2; ++t)
#pragma unroll
      for (int r = 0; r < 8; ++r)
        X[(size_t)(b * MAXA + q0 + rowAdd + r) * DIM + h * DKH + t * 16 + colBase] =
            (_Float16)x[t][r];
  }
}

// K6: h = X @ Wo^T + bo + residual(a), f32 out; residual gathered from a_flat
__global__ __launch_bounds__(256) void k_oproj(
    const _Float16* __restrict__ X, const _Float16* __restrict__ Wot,
    const float* __restrict__ bo, const float* __restrict__ a_flat,
    const int* __restrict__ a_sizes, const int* __restrict__ starts_a,
    float* __restrict__ Hout) {
  const int wave = threadIdx.x >> 5;
  const int lane = threadIdx.x & 31;
  const int m0 = blockIdx.x * 128 + wave * 16;
  const int n0 = blockIdx.y * 64;
  v8f acc[4];
#pragma unroll
  for (int t = 0; t < 4; ++t) {
    float bv = bo[n0 + t * 16 + (lane & 15)];
#pragma unroll
    for (int r = 0; r < 8; ++r) acc[t][r] = bv;
  }
  for (int k0 = 0; k0 < DIM; k0 += 32) {
    v16h af = load_fragA(X + (size_t)m0 * DIM + k0, DIM);
#pragma unroll
    for (int t = 0; t < 4; ++t) {
      v16h bf = load_fragB(Wot + (size_t)(n0 + t * 16) * DIM + k0, DIM);
      acc[t] = wmma16x16x32(af, bf, acc[t]);
    }
  }
  const int colBase = lane & 15;
  const int rowAdd  = (lane >> 4) << 3;
  const int b  = m0 >> 8;                               // 16 | 256: one graph/tile
  const int na = a_sizes[b];
  const int s0 = starts_a[b];
#pragma unroll
  for (int r = 0; r < 8; ++r) {
    const int row = m0 + rowAdd + r;
    const int q = row & 255;
    const bool qv = q < na;
    const float* ar = a_flat + (size_t)(s0 + q) * DIM;
#pragma unroll
    for (int t = 0; t < 4; ++t) {
      int col = n0 + t * 16 + colBase;
      float res = qv ? ar[col] : 0.0f;
      Hout[(size_t)row * DIM + col] = acc[t][r] + res;
    }
  }
}

// K7: tuple tail outputs: valid_a [NB*256], a_sizes [NB]
__global__ void k_tail(const int* __restrict__ a_sizes,
                       float* __restrict__ out_valid, float* __restrict__ out_sizes) {
  int idx = blockIdx.x * blockDim.x + threadIdx.x;
  if (idx < NB * MAXA) {
    int b = idx >> 8, q = idx & 255;
    out_valid[idx] = (q < a_sizes[b]) ? 1.0f : 0.0f;
  }
  if (idx < NB) out_sizes[idx] = (float)a_sizes[idx];
}

// ---------------------------------------------------------------------------
extern "C" void kernel_launch(void* const* d_in, const int* in_sizes, int n_in,
                              void* d_out, int out_size, void* d_ws, size_t ws_size,
                              hipStream_t stream) {
  const float* a_flat = (const float*)d_in[0];
  const float* p_flat = (const float*)d_in[1];
  const float* Wq = (const float*)d_in[2];  const float* bq = (const float*)d_in[3];
  const float* Wk = (const float*)d_in[4];  const float* bk = (const float*)d_in[5];
  const float* Wv = (const float*)d_in[6];  const float* bv = (const float*)d_in[7];
  const float* Wo = (const float*)d_in[8];  const float* bo = (const float*)d_in[9];
  const int* a_sizes = (const int*)d_in[10];
  const int* p_sizes = (const int*)d_in[11];

  size_t off = 0;
  auto alloc = [&](size_t bytes) {
    void* p = (char*)d_ws + off;
    off += (bytes + 255) & ~(size_t)255;
    return p;
  };
  int* starts_a = (int*)alloc(NB * sizeof(int));
  int* starts_p = (int*)alloc(NB * sizeof(int));
  _Float16* Wq_t = (_Float16*)alloc((size_t)DIM * DIM * 2);
  _Float16* Wk_t = (_Float16*)alloc((size_t)DIM * DIM * 2);
  _Float16* Wv_t = (_Float16*)alloc((size_t)DIM * DIM * 2);
  _Float16* Wo_t = (_Float16*)alloc((size_t)DIM * DIM * 2);
  _Float16* a_h  = (_Float16*)alloc((size_t)NB * MAXA * DIM * 2);
  _Float16* p_h  = (_Float16*)alloc((size_t)NB * MAXP * DIM * 2);
  _Float16* Qb   = (_Float16*)alloc((size_t)NB * MAXA * DIM * 2);
  _Float16* Kb   = (_Float16*)alloc((size_t)NB * MAXP * DIM * 2);
  _Float16* Vb   = (_Float16*)alloc((size_t)NB * MAXP * DIM * 2);
  _Float16* Vtb  = (_Float16*)alloc((size_t)NB * NH * DKH * MAXP * 2);
  _Float16* Xb   = (_Float16*)alloc((size_t)NB * MAXA * DIM * 2);

  float* out_h     = (float*)d_out;
  float* out_valid = out_h + (size_t)NB * MAXA * DIM;
  float* out_sizes = out_valid + (size_t)NB * MAXA;

  k_prefix<<<1, 32, 0, stream>>>(a_sizes, p_sizes, starts_a, starts_p);

  k_wt<<<DIM * DIM / 256, 256, 0, stream>>>(Wq, Wq_t);
  k_wt<<<DIM * DIM / 256, 256, 0, stream>>>(Wk, Wk_t);
  k_wt<<<DIM * DIM / 256, 256, 0, stream>>>(Wv, Wv_t);
  k_wt<<<DIM * DIM / 256, 256, 0, stream>>>(Wo, Wo_t);

  k_gather<<<NB * MAXA, 128, 0, stream>>>(a_flat, a_sizes, starts_a, a_h, MAXA);
  k_gather<<<NB * MAXP, 128, 0, stream>>>(p_flat, p_sizes, starts_p, p_h, MAXP);

  const int Ma = NB * MAXA, Mp = NB * MAXP;
  k_gemm_f16out<<<dim3(Ma / 128, DIM / 64), 256, 0, stream>>>(a_h, Wq_t, bq, Qb, Ma);
  k_gemm_f16out<<<dim3(Mp / 128, DIM / 64), 256, 0, stream>>>(p_h, Wk_t, bk, Kb, Mp);
  k_gemm_f16out<<<dim3(Mp / 128, DIM / 64), 256, 0, stream>>>(p_h, Wv_t, bv, Vb, Mp);

  k_vt<<<(NB * MAXP * DIM) / 256, 256, 0, stream>>>(Vb, Vtb);

  k_attn<<<NB * NH, 256, 0, stream>>>(Qb, Kb, Vtb, a_sizes, p_sizes, Xb);

  k_oproj<<<dim3(Ma / 128, DIM / 64), 256, 0, stream>>>(Xb, Wo_t, bo, a_flat,
                                                        a_sizes, starts_a, out_h);

  k_tail<<<(NB * MAXA + 255) / 256, 256, 0, stream>>>(a_sizes, out_valid, out_sizes);
}
